// GCN_3882650436604
// MI455X (gfx1250) — matrix operand order repaired and checked by reading, not compiled, not run on hardware
//
#include <hip/hip_runtime.h>

typedef __attribute__((ext_vector_type(16))) __bf16 v16bf;
typedef __attribute__((ext_vector_type(4)))  __bf16 v4bf;
typedef __attribute__((ext_vector_type(8)))  float  v8f;
typedef __attribute__((ext_vector_type(4)))  float  v4f;

__device__ __forceinline__ void atomic_add_f32(float* p, float v) {
    __hip_atomic_fetch_add(p, v, __ATOMIC_RELAXED, __HIP_MEMORY_SCOPE_AGENT);
}

// ---------------- zero workspace ----------------
__global__ void k_zero(float* __restrict__ p, size_t n) {
    size_t i4 = ((size_t)blockIdx.x * blockDim.x + threadIdx.x) * 4;
    if (i4 + 3 < n) {
        v4f z = {0.f, 0.f, 0.f, 0.f};
        *(v4f*)(p + i4) = z;
    } else {
        for (size_t j = i4; j < n; ++j) p[j] = 0.f;
    }
}

// ---------------- pre-swizzle weights into WMMA B-fragment layout ----------------
// B fragment (16-bit, KxN=32x16): lane L holds column n=L&15; elements e=0..15 hold
// K = (L>>4)*8 + e for e<8, 16 + (L>>4)*8 + (e-8) for e>=8. Fragment f = kc*nct + ct stored as 32 lanes x 16 bf16 contiguous.
__global__ void k_swizzleB(const float* __restrict__ W, __bf16* __restrict__ Ws,
                           int kchunks, int ldw, int ncols, int nct) {
    int t = blockIdx.x * blockDim.x + threadIdx.x;
    int nfrag = kchunks * nct;
    if (t >= nfrag * 32) return;
    int frag = t >> 5, lane = t & 31;
    int kc = frag / nct, ct = frag % nct;
    int n = lane & 15, kh = (lane >> 4) * 16;
    int col = ct * 16 + n;
    __bf16* out = Ws + (size_t)t * 16;
    for (int e = 0; e < 16; ++e) {
        float v = (col < ncols) ? W[(size_t)(kc * 32 + (kh >> 1) + (e & 7) + ((e >> 3) << 4)) * ldw + col] : 0.f;
        out[e] = (__bf16)v;
    }
}

// ---------------- GEMM1: h1[N,128](bf16) = x[N,512](f32) @ W1s ----------------
// block = 256 threads = 8 waves; wave w owns rows [blockIdx*128 + w*16, +16), all 128 cols.
__global__ __launch_bounds__(256) void k_gemm1(const float* __restrict__ x,
                                               const __bf16* __restrict__ W1s,
                                               __bf16* __restrict__ h1, int N) {
    const int lane = threadIdx.x & 31;
    const int wave = threadIdx.x >> 5;
    const int row0 = blockIdx.x * 128 + wave * 16;
    if (row0 >= N) return;                        // wave-uniform exit (N % 16 == 0)
    const int m    = lane & 15;
    const int halo = (lane >> 4) * 8;             // A-matrix K base for this lane half
    const float* xrow = x + (size_t)(row0 + m) * 512;

    v8f zero8 = {0.f,0.f,0.f,0.f,0.f,0.f,0.f,0.f};
    v8f acc[8];
#pragma unroll
    for (int ct = 0; ct < 8; ++ct) acc[ct] = zero8;

    // double-buffered B fragments: load frag(ct+1) before WMMA consuming frag(ct)
    v16bf bcur = *(const v16bf*)(W1s + (size_t)lane * 16);   // kc=0, ct=0
    for (int kc = 0; kc < 16; ++kc) {
        const int k0 = kc * 32;
        v4f a0 = *(const v4f*)(xrow + k0 + halo + 0);
        v4f a1 = *(const v4f*)(xrow + k0 + halo + 4);
        v4f a2 = *(const v4f*)(xrow + k0 + halo + 16);
        v4f a3 = *(const v4f*)(xrow + k0 + halo + 20);
        v16bf a;
#pragma unroll
        for (int j = 0; j < 4; ++j) {
            a[j]      = (__bf16)a0[j];
            a[4 + j]  = (__bf16)a1[j];
            a[8 + j]  = (__bf16)a2[j];
            a[12 + j] = (__bf16)a3[j];
        }
#pragma unroll
        for (int ct = 0; ct < 8; ++ct) {
            // next fragment index (clamped; value unused on the very last step)
            int nf = kc * 8 + ct + 1;
            if (nf > 127) nf = 127;
            v16bf bnxt = *(const v16bf*)(W1s + ((size_t)nf * 32 + lane) * 16);
            acc[ct] = __builtin_amdgcn_wmma_f32_16x16x32_bf16(
                false, a, false, bcur, (short)0, acc[ct], false, false);
            bcur = bnxt;
        }
    }
    // C/D layout: lane L holds col n=L&15; VGPR r is row m = r + 8*(L>>4)
#pragma unroll
    for (int ct = 0; ct < 8; ++ct) {
        const int col = ct * 16 + (lane & 15);
#pragma unroll
        for (int r = 0; r < 8; ++r) {
            int row = row0 + r + 8 * (lane >> 4);
            h1[(size_t)row * 128 + col] = (__bf16)acc[ct][r];
        }
    }
}

// ---------------- spmm1: hacc[dst] += val * h1[src], F=128 ----------------
// one wave per edge; lane handles 4 features (8B bf16x4 gather, 4 f32 atomics)
__global__ __launch_bounds__(256) void k_spmm1(const int* __restrict__ esrc,
                                               const int* __restrict__ edst,
                                               const float* __restrict__ eval,
                                               const __bf16* __restrict__ h1,
                                               float* __restrict__ hacc, int E) {
    int e = blockIdx.x * 8 + (threadIdx.x >> 5);
    if (e >= E) return;
    int lane = threadIdx.x & 31;
    int s = esrc[e], d = edst[e];
    float v = eval[e];
    v4bf hv = *(const v4bf*)(h1 + (size_t)s * 128 + lane * 4);
    float* op = hacc + (size_t)d * 128 + lane * 4;
    atomic_add_f32(op + 0, v * (float)hv[0]);
    atomic_add_f32(op + 1, v * (float)hv[1]);
    atomic_add_f32(op + 2, v * (float)hv[2]);
    atomic_add_f32(op + 3, v * (float)hv[3]);
}

// ---------------- GEMM2: s2[N,40](f32) = relu(hacc + b1) @ W2s (cols padded to 48) ----
// bias + ReLU fused into the A-matrix f32->bf16 conversion (K coord = k0+halo+e for e<8, k0+halo+8+e for e>=8).
__global__ __launch_bounds__(256) void k_gemm2(const float* __restrict__ hacc,
                                               const float* __restrict__ b1,
                                               const __bf16* __restrict__ W2s,
                                               float* __restrict__ s2, int N) {
    const int lane = threadIdx.x & 31;
    const int wave = threadIdx.x >> 5;
    const int row0 = blockIdx.x * 128 + wave * 16;
    if (row0 >= N) return;
    const int m    = lane & 15;
    const int halo = (lane >> 4) * 8;
    const float* hrow = hacc + (size_t)(row0 + m) * 128;

    v8f zero8 = {0.f,0.f,0.f,0.f,0.f,0.f,0.f,0.f};
    v8f acc[3];
#pragma unroll
    for (int ct = 0; ct < 3; ++ct) acc[ct] = zero8;

    v16bf bcur = *(const v16bf*)(W2s + (size_t)lane * 16);   // kc=0, ct=0
#pragma unroll
    for (int kc = 0; kc < 4; ++kc) {
        const int k0 = kc * 32;
        v4f a0 = *(const v4f*)(hrow + k0 + halo + 0);
        v4f a1 = *(const v4f*)(hrow + k0 + halo + 4);
        v4f a2 = *(const v4f*)(hrow + k0 + halo + 16);
        v4f a3 = *(const v4f*)(hrow + k0 + halo + 20);
        v4f c0 = *(const v4f*)(b1 + k0 + halo + 0);
        v4f c1 = *(const v4f*)(b1 + k0 + halo + 4);
        v4f c2 = *(const v4f*)(b1 + k0 + halo + 16);
        v4f c3 = *(const v4f*)(b1 + k0 + halo + 20);
        v16bf a;
#pragma unroll
        for (int j = 0; j < 4; ++j) {
            a[j]      = (__bf16)fmaxf(a0[j] + c0[j], 0.f);
            a[4 + j]  = (__bf16)fmaxf(a1[j] + c1[j], 0.f);
            a[8 + j]  = (__bf16)fmaxf(a2[j] + c2[j], 0.f);
            a[12 + j] = (__bf16)fmaxf(a3[j] + c3[j], 0.f);
        }
#pragma unroll
        for (int ct = 0; ct < 3; ++ct) {
            int nf = kc * 3 + ct + 1;
            if (nf > 11) nf = 11;
            v16bf bnxt = *(const v16bf*)(W2s + ((size_t)nf * 32 + lane) * 16);
            acc[ct] = __builtin_amdgcn_wmma_f32_16x16x32_bf16(
                false, a, false, bcur, (short)0, acc[ct], false, false);
            bcur = bnxt;
        }
    }
#pragma unroll
    for (int ct = 0; ct < 3; ++ct) {
        const int col = ct * 16 + (lane & 15);
        if (col < 40) {
#pragma unroll
            for (int r = 0; r < 8; ++r) {
                int row = row0 + r + 8 * (lane >> 4);
                s2[(size_t)row * 40 + col] = acc[ct][r];
            }
        }
    }
}

// ---------------- spmm2: oacc[dst] += val * s2[src], F=40 ----------------
__global__ __launch_bounds__(256) void k_spmm2(const int* __restrict__ esrc,
                                               const int* __restrict__ edst,
                                               const float* __restrict__ eval,
                                               const float* __restrict__ s2,
                                               float* __restrict__ oacc, int E) {
    int e = blockIdx.x * 8 + (threadIdx.x >> 5);
    if (e >= E) return;
    int lane = threadIdx.x & 31;
    int s = esrc[e], d = edst[e];
    float v = eval[e];
    float a = s2[(size_t)s * 40 + lane];
    atomic_add_f32(oacc + (size_t)d * 40 + lane, v * a);
    if (lane < 8) {
        float b = s2[(size_t)s * 40 + 32 + lane];
        atomic_add_f32(oacc + (size_t)d * 40 + 32 + lane, v * b);
    }
}

// ---------------- bias + log_softmax over 40 classes ----------------
__global__ void k_logsoftmax(const float* __restrict__ oacc, const float* __restrict__ b2,
                             float* __restrict__ out, int N) {
    int r = blockIdx.x * blockDim.x + threadIdx.x;
    if (r >= N) return;
    float vals[40];
    float mx = -3.402823466e+38f;
#pragma unroll
    for (int f = 0; f < 40; ++f) {
        float v = oacc[(size_t)r * 40 + f] + b2[f];
        vals[f] = v;
        mx = fmaxf(mx, v);
    }
    float s = 0.f;
#pragma unroll
    for (int f = 0; f < 40; ++f) s += __expf(vals[f] - mx);
    float lse = mx + __logf(s);
#pragma unroll
    for (int f = 0; f < 40; ++f) out[(size_t)r * 40 + f] = vals[f] - lse;
}

extern "C" void kernel_launch(void* const* d_in, const int* in_sizes, int n_in,
                              void* d_out, int out_size, void* d_ws, size_t ws_size,
                              hipStream_t stream) {
    const float* x    = (const float*)d_in[0];
    const int*   esrc = (const int*)d_in[1];
    const int*   edst = (const int*)d_in[2];
    const float* evalp= (const float*)d_in[3];
    const float* W1   = (const float*)d_in[4];
    const float* b1   = (const float*)d_in[5];
    const float* W2   = (const float*)d_in[6];
    const float* b2   = (const float*)d_in[7];
    float* out = (float*)d_out;

    const int N = in_sizes[0] / 512;   // 100000
    const int E = in_sizes[1];         // 3200000

    // workspace carve-up (256B aligned)
    char* ws = (char*)d_ws;
    size_t off = 0;
    auto carve = [&](size_t bytes) { char* p = ws + off; off += (bytes + 255) & ~(size_t)255; return p; };
    __bf16* W1s  = (__bf16*)carve((size_t)16 * 8 * 32 * 16 * 2);   // 128 KB
    __bf16* W2s  = (__bf16*)carve((size_t)4 * 3 * 32 * 16 * 2);    // 12 KB
    __bf16* h1   = (__bf16*)carve((size_t)N * 128 * 2);            // 25.6 MB
    float*  hacc = (float*)carve((size_t)N * 128 * 4);             // 51.2 MB
    float*  s2   = (float*)carve((size_t)N * 40 * 4);              // 16 MB
    float*  oacc = (float*)carve((size_t)N * 40 * 4);              // 16 MB

    const size_t nH = (size_t)N * 128, nO = (size_t)N * 40;

    // 1. zero accumulators
    k_zero<<<(unsigned)((nH / 4 + 255) / 256), 256, 0, stream>>>(hacc, nH);
    k_zero<<<(unsigned)((nO / 4 + 255) / 256), 256, 0, stream>>>(oacc, nO);
    // 2. swizzle weights into B-fragment layout
    k_swizzleB<<<(16 * 8 * 32 + 255) / 256, 256, 0, stream>>>(W1, W1s, 16, 128, 128, 8);
    k_swizzleB<<<(4 * 3 * 32 + 255) / 256, 256, 0, stream>>>(W2, W2s, 4, 40, 40, 3);
    // 3. h1 = bf16(x @ W1)
    int gblocks = (N + 127) / 128;
    k_gemm1<<<gblocks, 256, 0, stream>>>(x, W1s, h1, N);
    // 4. hacc = spmm(h1)
    k_spmm1<<<(E + 7) / 8, 256, 0, stream>>>(esrc, edst, evalp, h1, hacc, E);
    // 5. s2 = relu(hacc + b1) @ W2   (bias+ReLU fused into A conversion)
    k_gemm2<<<gblocks, 256, 0, stream>>>(hacc, b1, W2s, s2, N);
    // 6. oacc = spmm(s2)
    k_spmm2<<<(E + 7) / 8, 256, 0, stream>>>(esrc, edst, evalp, s2, oacc, E);
    // 7. out = log_softmax(oacc + b2)
    k_logsoftmax<<<(N + 255) / 256, 256, 0, stream>>>(oacc, b2, out, N);
}
